// BipartiteGraphConvNet_59785944760588
// MI455X (gfx1250) — matrix operand order repaired and checked by reading, not compiled, not run on hardware
//
#include <hip/hip_runtime.h>
#include <hip/hip_bf16.h>
#include <stdint.h>
#include <stddef.h>

// Problem sizes (fixed by the reference)
#define NN    8192      // n_uni == n_sub
#define DFEAT 128       // d_uni == d_sub
#define NF    (NN * DFEAT)
#define LDK   136       // padded LDS K-stride (elements): 272B rows -> conflict-free
#define KCH   128       // K-chunk staged per LDS buffer
#define NCH   (NN / KCH)

typedef __bf16 bf16;
typedef bf16  bf16x4  __attribute__((ext_vector_type(4)));
typedef bf16  bf16x8  __attribute__((ext_vector_type(8)));
typedef bf16  bf16x16 __attribute__((ext_vector_type(16)));
typedef float f32x4   __attribute__((ext_vector_type(4)));
typedef float f32x8   __attribute__((ext_vector_type(8)));

// ---------------------------------------------------------------------------
// CDNA5 async global->LDS copy (contiguous 16B per lane), tracked by ASYNCcnt.
// lds_addr: byte address within LDS (= low 32 bits of generic shared pointer).
// ---------------------------------------------------------------------------
__device__ __forceinline__ void async_ld_b128(unsigned lds_addr, const void* g) {
    asm volatile("global_load_async_to_lds_b128 %0, %1, off"
                 :: "v"(lds_addr), "v"(g) : "memory");
}
// Wait until at most N async ops remain outstanding (in-order completion).
#define WAIT_ASYNC(n) asm volatile("s_wait_asynccnt %0" :: "n"(n) : "memory")

// ---------------------------------------------------------------------------
// Pass 0a: adj (f32) -> adj (bf16) + row sums.  One block per row.
// ---------------------------------------------------------------------------
__global__ __launch_bounds__(256)
void k_adj_bf16_rowsum(const float* __restrict__ adj,
                       bf16* __restrict__ adjb,
                       float* __restrict__ rowsum) {
    const int r = blockIdx.x;
    const float* row = adj + (size_t)r * NN;
    bf16* brow = adjb + (size_t)r * NN;

    float s = 0.0f;
    for (int c = threadIdx.x * 4; c < NN; c += blockDim.x * 4) {
        f32x4 v = *(const f32x4*)(row + c);
        s += v.x + v.y + v.z + v.w;
        bf16x4 b = { (bf16)v.x, (bf16)v.y, (bf16)v.z, (bf16)v.w };
        *(bf16x4*)(brow + c) = b;
    }

    __shared__ float red[256];
    red[threadIdx.x] = s;
    __syncthreads();
    for (int off = 128; off > 0; off >>= 1) {
        if (threadIdx.x < off) red[threadIdx.x] += red[threadIdx.x + off];
        __syncthreads();
    }
    if (threadIdx.x == 0) rowsum[r] = red[0];
}

// ---------------------------------------------------------------------------
// Pass 0b: column sums of bf16 adj, deterministic two stage (no FP atomics).
// ---------------------------------------------------------------------------
#define CS_RBLK 512
__global__ __launch_bounds__(256)
void k_colsum_part(const bf16* __restrict__ adjb, float* __restrict__ part) {
    const int c  = blockIdx.x * 256 + threadIdx.x;
    const int r0 = blockIdx.y * CS_RBLK;
    float s = 0.0f;
    for (int r = r0; r < r0 + CS_RBLK; ++r)
        s += (float)adjb[(size_t)r * NN + c];
    part[(size_t)blockIdx.y * NN + c] = s;
}

__global__ __launch_bounds__(256)
void k_colsum_reduce(const float* __restrict__ part, float* __restrict__ colsum) {
    const int c = blockIdx.x * 256 + threadIdx.x;
    float s = 0.0f;
    #pragma unroll
    for (int b = 0; b < NN / CS_RBLK; ++b)
        s += part[(size_t)b * NN + c];
    colsum[c] = s;
}

// ---------------------------------------------------------------------------
// f32 [NN,DFEAT] -> bf16 K-transposed [DFEAT][NN]
// ---------------------------------------------------------------------------
__global__ __launch_bounds__(256)
void k_f2bf_T(const float* __restrict__ in, bf16* __restrict__ outT) {
    int i = blockIdx.x * blockDim.x + threadIdx.x;   // index into outT
    if (i < NF) {
        int d = i >> 13;        // / NN
        int r = i & (NN - 1);   // % NN
        outT[i] = (bf16)in[(size_t)r * DFEAT + d];
    }
}

// ---------------------------------------------------------------------------
// WMMA fragment helpers for v_wmma_f32_16x16x32_bf16 (wave32 layouts, ISA 7.12.2)
// ---------------------------------------------------------------------------
__device__ __forceinline__ bf16x16 cat8(bf16x8 lo, bf16x8 hi) {
    return __builtin_shufflevector(lo, hi, 0, 1, 2, 3, 4, 5, 6, 7,
                                           8, 9, 10, 11, 12, 13, 14, 15);
}

// A operand 16x32 (MxK), row-major source, leading dim ld (elements).
// lane (mr = lane&15, h = lane>>4) holds k in [8h,8h+8) U [8h+16,8h+24).
__device__ __forceinline__ bf16x16 afrag(const bf16* base, int ld) {
    const int lane = threadIdx.x & 31;
    const int mr = lane & 15, h = lane >> 4;
    const bf16* p = base + (size_t)mr * ld + 8 * h;
    return cat8(*(const bf16x8*)p, *(const bf16x8*)(p + 16));
}

// B operand 32x16 (KxN) from a K-transposed tile T[n][k] (stride ld elements).
// lane (n = lane&15 + 16*ntile, h = lane>>4) holds k in [k0+8h,k0+8h+8) U [k0+8h+16,k0+8h+24).
__device__ __forceinline__ bf16x16 bfrag(const bf16* T, int ld, int ntile, int k0) {
    const int lane = threadIdx.x & 31;
    const int n = (lane & 15) + 16 * ntile;
    const int h = lane >> 4;
    const bf16* p = T + (size_t)n * ld + k0 + 8 * h;
    return cat8(*(const bf16x8*)p, *(const bf16x8*)(p + 16));
}

__device__ __forceinline__ f32x8 wmma_bf16(bf16x16 a, bf16x16 b, f32x8 c) {
    return __builtin_amdgcn_wmma_f32_16x16x32_bf16(false, a, false, b,
                                                   (short)0, c, false, false);
}

// ---------------------------------------------------------------------------
// mid[i,:] = (1/rowsum[i]) * sum_k A[i,k] * XT[:,k]^T
// A row-major bf16 [NN,NN]; XT K-transposed bf16 [DFEAT][NN].
// Block: 64 threads (2 waves); each wave computes 32 rows (two M-tiles) x 128
// cols. X tile double-buffered in LDS via async global->LDS copies; next chunk
// streams while current chunk is consumed (partial s_wait_asynccnt).
// ---------------------------------------------------------------------------
__global__ __launch_bounds__(64, 1)
void k_gemmN(const bf16* __restrict__ A, const bf16* __restrict__ XT,
             const float* __restrict__ rsum, float* __restrict__ mid) {
    __shared__ __align__(16) bf16 Xs[2][DFEAT][LDK];   // [buf][n][k]

    const int tid  = threadIdx.x;
    const int wave = tid >> 5;
    const int lane = tid & 31;
    const int mrow = blockIdx.x * 64 + 32 * wave;
    const bf16* A0 = A + (size_t)mrow * NN;
    const bf16* A1 = A + (size_t)(mrow + 16) * NN;

    f32x8 acc[2][8];
    #pragma unroll
    for (int g = 0; g < 2; ++g)
        #pragma unroll
        for (int t = 0; t < 8; ++t) acc[g][t] = (f32x8){0, 0, 0, 0, 0, 0, 0, 0};

    // Stage chunk 0 into buffer 0 (32 async instructions per wave).
    {
        const unsigned base = (unsigned)(uintptr_t)&Xs[0][0][0];
        for (int c = tid; c < 2048; c += 64) {
            int n = c >> 4, ck = c & 15;
            async_ld_b128(base + (unsigned)((n * LDK + ck * 8) * 2),
                          XT + (size_t)n * NN + ck * 8);
        }
    }

    for (int ich = 0; ich < NCH; ++ich) {
        const int k0 = ich * KCH;
        if (ich + 1 < NCH) {     // prefetch next chunk into other buffer
            const unsigned base = (unsigned)(uintptr_t)&Xs[(ich + 1) & 1][0][0];
            for (int c = tid; c < 2048; c += 64) {
                int n = c >> 4, ck = c & 15;
                async_ld_b128(base + (unsigned)((n * LDK + ck * 8) * 2),
                              XT + (size_t)n * NN + (k0 + KCH) + ck * 8);
            }
            WAIT_ASYNC(32);      // chunk ich has landed; chunk ich+1 in flight
        } else {
            WAIT_ASYNC(0);
        }
        __syncthreads();

        const bf16* xsb = &Xs[ich & 1][0][0];
        __builtin_prefetch(A0 + k0 + KCH, 0, 1);
        #pragma unroll
        for (int kk = 0; kk < KCH; kk += 32) {
            bf16x16 a0 = afrag(A0 + k0 + kk, NN);
            bf16x16 a1 = afrag(A1 + k0 + kk, NN);
            bf16x16 bc = bfrag(xsb, LDK, 0, kk);
            #pragma unroll
            for (int t = 0; t < 8; ++t) {
                bf16x16 bn;
                if (t < 7) bn = bfrag(xsb, LDK, t + 1, kk);  // overlap next B load
                acc[0][t] = wmma_bf16(a0, bc, acc[0][t]);
                acc[1][t] = wmma_bf16(a1, bc, acc[1][t]);
                if (t < 7) bc = bn;
            }
        }
        __syncthreads();
    }

    const int h = lane >> 4, nl = lane & 15;
    #pragma unroll
    for (int g = 0; g < 2; ++g) {
        const int rb = mrow + 16 * g;
        float inv[8];
        #pragma unroll
        for (int v = 0; v < 8; ++v) inv[v] = 1.0f / rsum[rb + 8 * h + v];
        #pragma unroll
        for (int t = 0; t < 8; ++t)
            #pragma unroll
            for (int v = 0; v < 8; ++v)
                mid[(size_t)(rb + 8 * h + v) * DFEAT + 16 * t + nl] =
                    acc[g][t][v] * inv[v];
    }
}

// ---------------------------------------------------------------------------
// mid[j,:] = (1/colsum[j]) * sum_k A[k,j] * XT[:,k]^T      (A^T GEMM)
// A tile transposed through LDS (double-buffered); X tile async double-buffered.
// ---------------------------------------------------------------------------
__global__ __launch_bounds__(64, 1)
void k_gemmT(const bf16* __restrict__ A, const bf16* __restrict__ XT,
             const float* __restrict__ csum, float* __restrict__ mid) {
    __shared__ __align__(16) bf16 At[2][64][LDK];      // [buf][j][k]
    __shared__ __align__(16) bf16 Xs[2][DFEAT][LDK];   // [buf][n][k]

    const int tid  = threadIdx.x;
    const int wave = tid >> 5;
    const int lane = tid & 31;
    const int j0   = blockIdx.x * 64;
    const int jrow = j0 + 32 * wave;

    f32x8 acc[2][8];
    #pragma unroll
    for (int g = 0; g < 2; ++g)
        #pragma unroll
        for (int t = 0; t < 8; ++t) acc[g][t] = (f32x8){0, 0, 0, 0, 0, 0, 0, 0};

    // Stage chunk 0.
    {
        const unsigned base = (unsigned)(uintptr_t)&Xs[0][0][0];
        for (int c = tid; c < 2048; c += 64) {
            int n = c >> 4, ck = c & 15;
            async_ld_b128(base + (unsigned)((n * LDK + ck * 8) * 2),
                          XT + (size_t)n * NN + ck * 8);
        }
        for (int idx = tid; idx < (KCH * 64) / 2; idx += 64) {
            int k = idx >> 5, j = (idx & 31) * 2;
            union { unsigned u; bf16 b[2]; } cv;
            cv.u = *(const unsigned*)(A + (size_t)k * NN + j0 + j);
            At[0][j][k]     = cv.b[0];
            At[0][j + 1][k] = cv.b[1];
        }
    }

    for (int ich = 0; ich < NCH; ++ich) {
        const int k0 = ich * KCH;
        if (ich + 1 < NCH) {
            const int nb = (ich + 1) & 1;
            const unsigned base = (unsigned)(uintptr_t)&Xs[nb][0][0];
            for (int c = tid; c < 2048; c += 64) {
                int n = c >> 4, ck = c & 15;
                async_ld_b128(base + (unsigned)((n * LDK + ck * 8) * 2),
                              XT + (size_t)n * NN + (k0 + KCH) + ck * 8);
            }
            for (int idx = tid; idx < (KCH * 64) / 2; idx += 64) {
                int k = idx >> 5, j = (idx & 31) * 2;
                union { unsigned u; bf16 b[2]; } cv;
                cv.u = *(const unsigned*)(A + (size_t)(k0 + KCH + k) * NN + j0 + j);
                At[nb][j][k]     = cv.b[0];
                At[nb][j + 1][k] = cv.b[1];
            }
            WAIT_ASYNC(32);
        } else {
            WAIT_ASYNC(0);
        }
        __syncthreads();

        const int cb = ich & 1;
        const bf16* xsb = &Xs[cb][0][0];
        #pragma unroll
        for (int kk = 0; kk < KCH; kk += 32) {
            bf16x16 a0 = afrag(&At[cb][32 * wave][kk], LDK);
            bf16x16 a1 = afrag(&At[cb][32 * wave + 16][kk], LDK);
            bf16x16 bc = bfrag(xsb, LDK, 0, kk);
            #pragma unroll
            for (int t = 0; t < 8; ++t) {
                bf16x16 bn;
                if (t < 7) bn = bfrag(xsb, LDK, t + 1, kk);
                acc[0][t] = wmma_bf16(a0, bc, acc[0][t]);
                acc[1][t] = wmma_bf16(a1, bc, acc[1][t]);
                if (t < 7) bc = bn;
            }
        }
        __syncthreads();
    }

    const int h = lane >> 4, nl = lane & 15;
    #pragma unroll
    for (int g = 0; g < 2; ++g) {
        const int rb = jrow + 16 * g;
        float inv[8];
        #pragma unroll
        for (int v = 0; v < 8; ++v) inv[v] = 1.0f / csum[rb + 8 * h + v];
        #pragma unroll
        for (int t = 0; t < 8; ++t)
            #pragma unroll
            for (int v = 0; v < 8; ++v)
                mid[(size_t)(rb + 8 * h + v) * DFEAT + 16 * t + nl] =
                    acc[g][t][v] * inv[v];
    }
}

// ---------------------------------------------------------------------------
// out[r,:] = mid[r,:] @ W   (mid [NN,128] f32, W [128,128] f32, K = 128)
// Writes f32 (final outputs) and/or K-transposed bf16 (next-pass inputs).
// ---------------------------------------------------------------------------
__global__ __launch_bounds__(128)
void k_wgemm(const float* __restrict__ mid, const float* __restrict__ W,
             float* __restrict__ outf, bf16* __restrict__ outbT) {
    __shared__ __align__(16) bf16 WsT[DFEAT][LDK];  // W^T: [d][c]

    const int tid  = threadIdx.x;
    const int wave = tid >> 5;
    const int lane = tid & 31;
    const int mrow = blockIdx.x * 64 + 16 * wave;
    const int mr = lane & 15, h = lane >> 4;

    for (int idx = tid; idx < DFEAT * DFEAT; idx += 128) {
        int c = idx >> 7;
        int d = idx & 127;
        WsT[d][c] = (bf16)W[idx];
    }

    f32x8 acc[8];
    #pragma unroll
    for (int t = 0; t < 8; ++t) acc[t] = (f32x8){0, 0, 0, 0, 0, 0, 0, 0};
    __syncthreads();

    #pragma unroll
    for (int c0 = 0; c0 < DFEAT; c0 += 32) {
        const float* p = mid + (size_t)(mrow + mr) * DFEAT + c0 + 8 * h;
        bf16x16 a;
        #pragma unroll
        for (int e = 0; e < 8; ++e) {
            a[e]     = (bf16)p[e];        // k = 8h + e
            a[8 + e] = (bf16)p[16 + e];   // k = 8h + 16 + e
        }
        bf16x16 bc = bfrag(&WsT[0][0], LDK, 0, c0);
        #pragma unroll
        for (int t = 0; t < 8; ++t) {
            bf16x16 bn;
            if (t < 7) bn = bfrag(&WsT[0][0], LDK, t + 1, c0);
            acc[t] = wmma_bf16(a, bc, acc[t]);
            if (t < 7) bc = bn;
        }
    }

    const int nl = lane & 15;
    #pragma unroll
    for (int t = 0; t < 8; ++t) {
        #pragma unroll
        for (int v = 0; v < 8; ++v) {
            int m = mrow + 8 * h + v;
            int n = 16 * t + nl;
            float val = acc[t][v];
            if (outf)  outf[(size_t)m * DFEAT + n] = val;
            if (outbT) outbT[(size_t)n * NN + m] = (bf16)val;   // K-transposed
        }
    }
}

// ---------------------------------------------------------------------------
// Host-side orchestration
// ---------------------------------------------------------------------------
extern "C" void kernel_launch(void* const* d_in, const int* in_sizes, int n_in,
                              void* d_out, int out_size, void* d_ws, size_t ws_size,
                              hipStream_t stream) {
    (void)in_sizes; (void)n_in; (void)out_size; (void)ws_size;

    const float* adj = (const float*)d_in[0];   // [NN, NN]
    const float* uni = (const float*)d_in[1];   // [NN, 128]
    const float* sub = (const float*)d_in[2];   // [NN, 128]
    const float* ws1 = (const float*)d_in[3];   // [128, 128]
    const float* wu1 = (const float*)d_in[4];   // [128, 128]
    float* out = (float*)d_out;                 // [sub_out | uni_out]

    char* wp = (char*)d_ws;
    auto take = [&](size_t bytes) -> void* {
        void* q = (void*)wp;
        wp += (bytes + 255) & ~(size_t)255;
        return q;
    };
    bf16*  adjb    = (bf16*)take((size_t)NN * NN * sizeof(bf16));   // 128 MB (fits 192 MB L2)
    float* rowsum  = (float*)take((size_t)NN * sizeof(float));
    float* colsum  = (float*)take((size_t)NN * sizeof(float));
    float* cspart  = (float*)take((size_t)(NN / CS_RBLK) * NN * sizeof(float));
    bf16*  Ub[2]   = { (bf16*)take((size_t)NF * sizeof(bf16)),      // K-transposed [128][NN]
                       (bf16*)take((size_t)NF * sizeof(bf16)) };
    bf16*  Sb[2]   = { (bf16*)take((size_t)NF * sizeof(bf16)),
                       (bf16*)take((size_t)NF * sizeof(bf16)) };
    float* midS    = (float*)take((size_t)NF * sizeof(float));
    float* midU    = (float*)take((size_t)NF * sizeof(float));

    // Precompute: bf16 adj + row sums + col sums, K-transposed bf16 features.
    k_adj_bf16_rowsum<<<NN, 256, 0, stream>>>(adj, adjb, rowsum);
    k_colsum_part<<<dim3(NN / 256, NN / CS_RBLK), 256, 0, stream>>>(adjb, cspart);
    k_colsum_reduce<<<NN / 256, 256, 0, stream>>>(cspart, colsum);
    k_f2bf_T<<<(NF + 255) / 256, 256, 0, stream>>>(uni, Ub[0]);
    k_f2bf_T<<<(NF + 255) / 256, 256, 0, stream>>>(sub, Sb[0]);

    // Three fused propagation passes:
    //   midS = diag(1/colsum) * (A^T @ U) ; S_next = midS @ w_s1
    //   midU = diag(1/rowsum) * (A   @ S) ; U_next = midU @ w_u1
    int cur = 0;
    for (int pass = 0; pass < 3; ++pass) {
        const bool last = (pass == 2);
        k_gemmT<<<NN / 64, 64, 0, stream>>>(adjb, Ub[cur], colsum, midS);
        k_gemmN<<<NN / 64, 64, 0, stream>>>(adjb, Sb[cur], rowsum, midU);
        k_wgemm<<<NN / 64, 128, 0, stream>>>(midS, ws1,
                                             last ? out : nullptr,
                                             last ? nullptr : Sb[cur ^ 1]);
        k_wgemm<<<NN / 64, 128, 0, stream>>>(midU, wu1,
                                             last ? out + (size_t)NF : nullptr,
                                             last ? nullptr : Ub[cur ^ 1]);
        cur ^= 1;
    }
}
